// SchNetInteraction_3461743641022
// MI455X (gfx1250) — compile-verified, matrix operand order checked by reading, not run on hardware
//
#include <hip/hip_runtime.h>
#include <math.h>

// ---------------------------------------------------------------------------
// SchNet interaction block for MI455X (gfx1250, wave32, WMMA bf16 path).
//   h   = x @ w1 + b1
//   msg = [sin(r*w), cos(r*w), edge_attr, h[src]]        (288 features)
//   m   = gelu(msg@mw1+mb1); m = gelu(m@mw2+mb2); m = m@mw3+mb3
//   hsum= h + segment_sum(m, dst)
//   out = relu(hsum@w2+b2) @ w3 + b3
// GEMMs: bf16 inputs, fp32 accumulate via v_wmma_f32_16x16x32_bf16.
// All LDS tiles use conflict-free padded strides (136/152/296 elems -> row
// bank-steps 4/12/20 across the 16 fragment rows, 16B aligned for b128).
// Edge stage: persistent WGs, all 3 weight matrices transposed in bf16 LDS
// (~143KB; CDNA5 320KB/WGP), 64-edge tiles, 2 n-tiles per wave (A-fragment
// reuse), global_prefetch_b8 pipelining of next tile's loads.
// ---------------------------------------------------------------------------

typedef __bf16 bf16_t;
typedef __attribute__((ext_vector_type(16))) __bf16 v16bf;
typedef __attribute__((ext_vector_type(8)))  __bf16 v8bf;
typedef __attribute__((ext_vector_type(8)))  float  v8f;
typedef __attribute__((ext_vector_type(4)))  float  v4f;

#define THREADS 512   // 16 waves

__device__ __forceinline__ bf16_t to_bf16(float f) { return (bf16_t)f; }

__device__ __forceinline__ v8bf cvt8(v4f a, v4f b) {
  v8bf r;
  r[0] = (bf16_t)a[0]; r[1] = (bf16_t)a[1]; r[2] = (bf16_t)a[2]; r[3] = (bf16_t)a[3];
  r[4] = (bf16_t)b[0]; r[5] = (bf16_t)b[1]; r[6] = (bf16_t)b[2]; r[7] = (bf16_t)b[3];
  return r;
}

__device__ __forceinline__ v8f wmma_bf(v16bf a, v16bf b, v8f c) {
  return __builtin_amdgcn_wmma_f32_16x16x32_bf16(
      false, a, false, b, (short)0, c, false, false);
}

// One wave, one 16x16 tile. A [16 x 32*KB] (LDS, stride sA), B = W^T (stride sB).
// Contiguous k order, same in A and B: lane holds row/col (lane&15), K-half
// (lane>=16)*16; 16 contiguous bf16 = 32B per lane per K-block.
template <int KB>
__device__ __forceinline__ v8f mm16x16(const bf16_t* A, int sA, const bf16_t* Bt, int sB) {
  const int lane = threadIdx.x & 31;
  const int r  = lane & 15;
  const int hi = (lane >> 4) << 4;
  const bf16_t* ap = A  + r * sA + hi;
  const bf16_t* bp = Bt + r * sB + hi;
  v8f c = {};
#pragma unroll
  for (int kb = 0; kb < KB; ++kb)
    c = wmma_bf(*(const v16bf*)(ap + kb * 32), *(const v16bf*)(bp + kb * 32), c);
  return c;
}

// One wave, two n-tiles sharing the A fragment (halves A LDS traffic).
template <int KB>
__device__ __forceinline__ void mm16x16x2(const bf16_t* A, int sA,
                                          const bf16_t* Bt0, const bf16_t* Bt1, int sB,
                                          v8f& c0, v8f& c1) {
  const int lane = threadIdx.x & 31;
  const int r  = lane & 15;
  const int hi = (lane >> 4) << 4;
  const bf16_t* ap = A   + r * sA + hi;
  const bf16_t* b0 = Bt0 + r * sB + hi;
  const bf16_t* b1 = Bt1 + r * sB + hi;
  c0 = (v8f){}; c1 = (v8f){};
#pragma unroll
  for (int kb = 0; kb < KB; ++kb) {
    v16bf av  = *(const v16bf*)(ap + kb * 32);
    v16bf bv0 = *(const v16bf*)(b0 + kb * 32);
    v16bf bv1 = *(const v16bf*)(b1 + kb * 32);
    c0 = wmma_bf(av, bv0, c0);
    c1 = wmma_bf(av, bv1, c1);
  }
}

__device__ __forceinline__ float gelu_exact(float x) {
  return 0.5f * x * (1.0f + erff(x * 0.70710678118654752f));
}

// ---------------------------------------------------------------------------
// Kernel A: h = x @ w1 + b1 -> hsum (f32) and hb (bf16 gather copy)
// ---------------------------------------------------------------------------
#define WS1 136   // padded stride for 128-wide weight tiles (bank step 4)
#define AS1 152   // padded stride for 128-wide activation tiles (bank step 12)

__global__ void __launch_bounds__(THREADS)
k_node_linear1(const float* __restrict__ x, const float* __restrict__ w1,
               const float* __restrict__ b1, float* __restrict__ hsum,
               bf16_t* __restrict__ hb, int nNodes) {
  extern __shared__ char smem[];
  bf16_t* w1t = (bf16_t*)smem;                      // [128][136]  34816 B
  float*  b1s = (float*)(smem + 128 * WS1 * 2);     // [128]
  bf16_t* xt  = (bf16_t*)(smem + 128 * WS1 * 2 + 512); // [32][152]

  const int tid = threadIdx.x;
  for (int i = tid; i < 128 * 128; i += THREADS) {
    int o = i >> 7, k = i & 127;
    w1t[o * WS1 + k] = to_bf16(w1[k * 128 + o]);    // transpose -> [out][in]
  }
  if (tid < 128) b1s[tid] = b1[tid];
  __syncthreads();

  const int wave = tid >> 5, lane = tid & 31;
  const int me = wave >> 3, nt = wave & 7;
  const int col = nt * 16 + (lane & 15);
  const int hi8 = (lane >> 4) << 3;
  const int sr = tid >> 4, sc = (tid & 15) << 3;

  for (int tile = blockIdx.x; tile * 32 < nNodes; tile += gridDim.x) {
    const int row0 = tile * 32;
    const bool full = (row0 + 32 <= nNodes);

    {
      v8bf val = {};
      if (full || (row0 + sr) < nNodes) {
        const float* xr = x + (size_t)(row0 + sr) * 128 + sc;
        val = cvt8(*(const v4f*)xr, *(const v4f*)(xr + 4));
      }
      *(v8bf*)(xt + sr * AS1 + sc) = val;
    }
    __syncthreads();

    v8f c = mm16x16<4>(xt + me * 16 * AS1, AS1, w1t + nt * 16 * WS1, WS1);
    const float bias = b1s[col];
    const int row = row0 + me * 16 + hi8;
    if (full) {
      float*  hp = hsum + (size_t)row * 128 + col;
      bf16_t* hq = hb   + (size_t)row * 128 + col;
#pragma unroll
      for (int v = 0; v < 8; ++v) {
        float val = c[v] + bias;
        hp[(size_t)v * 128] = val;
        hq[(size_t)v * 128] = to_bf16(val);
      }
    } else {
#pragma unroll
      for (int v = 0; v < 8; ++v) {
        if (row + v < nNodes) {
          float val = c[v] + bias;
          hsum[(size_t)(row + v) * 128 + col] = val;
          hb[(size_t)(row + v) * 128 + col]   = to_bf16(val);
        }
      }
    }
    __syncthreads();
  }
}

// ---------------------------------------------------------------------------
// Kernel B: per-edge message MLP + fp32-atomic segment sum into hsum.
// 64-edge tiles; wave w: M-tile (w&3), n-pair (w>>2).
// ---------------------------------------------------------------------------
#define MS  296   // msg stride (288 padded; bank step 20)

__global__ void __launch_bounds__(THREADS)
k_edge_mlp(const int* __restrict__ ei, const float* __restrict__ ea,
           const float* __restrict__ xp,
           const float* __restrict__ mw1, const float* __restrict__ mb1,
           const float* __restrict__ mw2, const float* __restrict__ mb2,
           const float* __restrict__ mw3, const float* __restrict__ mb3,
           const bf16_t* __restrict__ hb, float* __restrict__ hsum, int E) {
  extern __shared__ char smem[];
  bf16_t* w1t  = (bf16_t*)smem;            // [128][296]  75776 B
  bf16_t* w2t  = w1t + 128 * MS;           // [128][136]  34816 B
  bf16_t* w3t  = w2t + 128 * WS1;          // [128][136]  34816 B
  float*  b1s  = (float*)(w3t + 128 * WS1);
  float*  b2s  = b1s + 128;
  float*  b3s  = b2s + 128;
  int*    dstL = (int*)(b3s + 128);        // [64]
  bf16_t* msg  = (bf16_t*)(dstL + 64);     // [64][296]  37888 B (16B-aligned)
  bf16_t* m1   = msg + 64 * MS;            // [64][152]  19456 B
  bf16_t* m2   = m1 + 64 * AS1;            // [64][152]  19456 B

  const int tid = threadIdx.x;
  // Stage transposed bf16 weights once per WG (amortized over ~16 tiles).
  for (int i = tid; i < 128 * 288; i += THREADS) {
    int o = i / 288, k = i % 288;
    w1t[o * MS + k] = to_bf16(mw1[k * 128 + o]);
  }
  for (int i = tid; i < 128 * 128; i += THREADS) {
    int o = i >> 7, k = i & 127;
    w2t[o * WS1 + k] = to_bf16(mw2[k * 128 + o]);
    w3t[o * WS1 + k] = to_bf16(mw3[k * 128 + o]);
  }
  if (tid < 128) { b1s[tid] = mb1[tid]; b2s[tid] = mb2[tid]; b3s[tid] = mb3[tid]; }
  __syncthreads();

  const int wave = tid >> 5, lane = tid & 31;
  const int me  = wave & 3;                 // M-tile 0..3 (16 edges each)
  const int nt0 = (wave >> 2) << 1;         // n-pair -> n-tiles nt0, nt0+1
  const int col0 = nt0 * 16 + (lane & 15);
  const int col1 = col0 + 16;
  const int hi8 = (lane >> 4) << 3;
  const int j = tid & 15;                   // staging: 16 threads per edge
  const float om = 10.f * exp2f(7.0f - 0.875f * (float)j);   // omega_j
  const int nTiles = (E + 63) >> 6;

  for (int tile = blockIdx.x; tile < nTiles; tile += gridDim.x) {
    const int e0 = tile * 64;
    const bool full = (e0 + 64 <= E);

    // --- build msg tile: [sin | cos | edge_attr | h[src]] as bf16 ---
#pragma unroll
    for (int it = 0; it < 2; ++it) {
      const int et = (tid >> 4) + (it << 5);          // 0..63
      const int e  = e0 + et;
      const int esafe = full ? e : (e < E ? e : E - 1);
      const bool valid = full || (e < E);
      const int s = ei[esafe];
      const int d = ei[E + esafe];
      if (j == 0) dstL[et] = d;
      float dx = xp[3 * d + 0] - xp[3 * s + 0];
      float dy = xp[3 * d + 1] - xp[3 * s + 1];
      float dz = xp[3 * d + 2] - xp[3 * s + 2];
      float feat = sqrtf(dx * dx + dy * dy + dz * dz) * om;
      msg[et * MS + j]      = valid ? to_bf16(__sinf(feat)) : (bf16_t)0.f;
      msg[et * MS + 16 + j] = valid ? to_bf16(__cosf(feat)) : (bf16_t)0.f;
      const float*  eaRow = ea + (size_t)esafe * 128 + (j << 3);
      const bf16_t* hRow  = hb + (size_t)s * 128 + (j << 3);
      v8bf eav = {}, hv = {};
      if (valid) {
        eav = cvt8(*(const v4f*)eaRow, *(const v4f*)(eaRow + 4));
        hv  = *(const v8bf*)hRow;
      }
      *(v8bf*)(msg + et * MS + 32  + (j << 3)) = eav;
      *(v8bf*)(msg + et * MS + 160 + (j << 3)) = hv;
    }

    // --- prefetch next tile while this tile's MLP runs ---
    {
      int tn = tile + gridDim.x;
      if (tid < 64 && tn < nTiles) {
        int en = tn * 64 + tid;
        if (en < E) {
          const float* p = ea + (size_t)en * 128;
          __builtin_prefetch(p, 0, 0);
          __builtin_prefetch(p + 64, 0, 0);
          int sn = ei[en];
          __builtin_prefetch(hb + (size_t)sn * 128, 0, 0);
        }
      }
    }
    __syncthreads();

    // --- layer 1: 16x288 @ 288x(2x16), GELU ---
    v8f c1a, c1b;
    mm16x16x2<9>(msg + me * 16 * MS, MS,
                 w1t + nt0 * 16 * MS, w1t + (nt0 + 1) * 16 * MS, MS, c1a, c1b);
    {
      const float ba = b1s[col0], bb = b1s[col1];
#pragma unroll
      for (int v = 0; v < 8; ++v) {
        bf16_t* row = m1 + (me * 16 + v + hi8) * AS1;
        row[col0] = to_bf16(gelu_exact(c1a[v] + ba));
        row[col1] = to_bf16(gelu_exact(c1b[v] + bb));
      }
    }
    __syncthreads();

    // --- layer 2: 16x128 @ 128x(2x16), GELU ---
    v8f c2a, c2b;
    mm16x16x2<4>(m1 + me * 16 * AS1, AS1,
                 w2t + nt0 * 16 * WS1, w2t + (nt0 + 1) * 16 * WS1, WS1, c2a, c2b);
    {
      const float ba = b2s[col0], bb = b2s[col1];
#pragma unroll
      for (int v = 0; v < 8; ++v) {
        bf16_t* row = m2 + (me * 16 + v + hi8) * AS1;
        row[col0] = to_bf16(gelu_exact(c2a[v] + ba));
        row[col1] = to_bf16(gelu_exact(c2b[v] + bb));
      }
    }
    __syncthreads();

    // --- layer 3: 16x128 @ 128x(2x16), atomic segment-sum scatter ---
    v8f c3a, c3b;
    mm16x16x2<4>(m2 + me * 16 * AS1, AS1,
                 w3t + nt0 * 16 * WS1, w3t + (nt0 + 1) * 16 * WS1, WS1, c3a, c3b);
    {
      const float ba = b3s[col0], bb = b3s[col1];
      if (full) {
#pragma unroll
        for (int v = 0; v < 8; ++v) {
          size_t base = (size_t)dstL[me * 16 + v + hi8] * 128;
          atomicAdd(&hsum[base + col0], c3a[v] + ba);
          atomicAdd(&hsum[base + col1], c3b[v] + bb);
        }
      } else {
#pragma unroll
        for (int v = 0; v < 8; ++v) {
          int er = me * 16 + v + hi8;
          if (e0 + er < E) {
            size_t base = (size_t)dstL[er] * 128;
            atomicAdd(&hsum[base + col0], c3a[v] + ba);
            atomicAdd(&hsum[base + col1], c3b[v] + bb);
          }
        }
      }
    }
    __syncthreads();
  }
}

// ---------------------------------------------------------------------------
// Kernel C: out = relu(hsum @ w2 + b2) @ w3 + b3
// ---------------------------------------------------------------------------
__global__ void __launch_bounds__(THREADS)
k_node_out(const float* __restrict__ hsum,
           const float* __restrict__ w2, const float* __restrict__ b2,
           const float* __restrict__ w3, const float* __restrict__ b3,
           float* __restrict__ out, int nNodes) {
  extern __shared__ char smem[];
  bf16_t* w2t = (bf16_t*)smem;               // [128][136]
  bf16_t* w3t = w2t + 128 * WS1;             // [128][136]
  float*  b2s = (float*)(w3t + 128 * WS1);
  float*  b3s = b2s + 128;
  bf16_t* ht  = (bf16_t*)(b3s + 128);        // [32][152]
  bf16_t* t1  = ht + 32 * AS1;               // [32][152]

  const int tid = threadIdx.x;
  for (int i = tid; i < 128 * 128; i += THREADS) {
    int o = i >> 7, k = i & 127;
    w2t[o * WS1 + k] = to_bf16(w2[k * 128 + o]);
    w3t[o * WS1 + k] = to_bf16(w3[k * 128 + o]);
  }
  if (tid < 128) { b2s[tid] = b2[tid]; b3s[tid] = b3[tid]; }
  __syncthreads();

  const int wave = tid >> 5, lane = tid & 31;
  const int me = wave >> 3, nt = wave & 7;
  const int col = nt * 16 + (lane & 15);
  const int hi8 = (lane >> 4) << 3;
  const int sr = tid >> 4, sc = (tid & 15) << 3;

  for (int tile = blockIdx.x; tile * 32 < nNodes; tile += gridDim.x) {
    const int row0 = tile * 32;
    const bool full = (row0 + 32 <= nNodes);

    {
      v8bf val = {};
      if (full || (row0 + sr) < nNodes) {
        const float* hr = hsum + (size_t)(row0 + sr) * 128 + sc;
        val = cvt8(*(const v4f*)hr, *(const v4f*)(hr + 4));
      }
      *(v8bf*)(ht + sr * AS1 + sc) = val;
    }
    __syncthreads();

    v8f c1 = mm16x16<4>(ht + me * 16 * AS1, AS1, w2t + nt * 16 * WS1, WS1);
    {
      const float bias = b2s[col];
#pragma unroll
      for (int v = 0; v < 8; ++v) {
        float xv = c1[v] + bias;
        t1[(me * 16 + v + hi8) * AS1 + col] = to_bf16(xv > 0.f ? xv : 0.f);
      }
    }
    __syncthreads();

    v8f c2 = mm16x16<4>(t1 + me * 16 * AS1, AS1, w3t + nt * 16 * WS1, WS1);
    {
      const float bias = b3s[col];
      const int row = row0 + me * 16 + hi8;
      if (full) {
        float* op = out + (size_t)row * 128 + col;
#pragma unroll
        for (int v = 0; v < 8; ++v)
          op[(size_t)v * 128] = c2[v] + bias;
      } else {
#pragma unroll
        for (int v = 0; v < 8; ++v)
          if (row + v < nNodes)
            out[(size_t)(row + v) * 128 + col] = c2[v] + bias;
      }
    }
    __syncthreads();
  }
}

// ---------------------------------------------------------------------------
extern "C" void kernel_launch(void* const* d_in, const int* in_sizes, int n_in,
                              void* d_out, int out_size, void* d_ws, size_t ws_size,
                              hipStream_t stream) {
  const float* x   = (const float*)d_in[0];
  const int*   ei  = (const int*)d_in[1];     // [2, E] (src row, dst row)
  const float* ea  = (const float*)d_in[2];
  const float* xp  = (const float*)d_in[3];
  const float* w1  = (const float*)d_in[4];
  const float* b1  = (const float*)d_in[5];
  const float* mw1 = (const float*)d_in[6];
  const float* mb1 = (const float*)d_in[7];
  const float* mw2 = (const float*)d_in[8];
  const float* mb2 = (const float*)d_in[9];
  const float* mw3 = (const float*)d_in[10];
  const float* mb3 = (const float*)d_in[11];
  const float* w2  = (const float*)d_in[12];
  const float* b2  = (const float*)d_in[13];
  const float* w3  = (const float*)d_in[14];
  const float* b3  = (const float*)d_in[15];

  const int nNodes = in_sizes[0] / 128;
  const int E      = in_sizes[1] / 2;

  float*  hsum = (float*)d_ws;                                       // [N,128] f32
  bf16_t* hbuf = (bf16_t*)((char*)d_ws + (size_t)nNodes * 128 * 4);  // [N,128] bf16

  const int nodeTiles = (nNodes + 31) / 32;

  const size_t shA = (size_t)(128 * WS1 + 32 * AS1) * 2 + 512;               // ~44 KB
  const size_t shB = (size_t)(128 * MS + 2 * 128 * WS1 + 64 * MS + 2 * 64 * AS1) * 2
                   + 3 * 128 * 4 + 64 * 4;                                   // ~219 KB
  const size_t shC = (size_t)(2 * 128 * WS1 + 2 * 32 * AS1) * 2 + 2 * 128 * 4; // ~88 KB

  hipFuncSetAttribute((const void*)k_edge_mlp,
                      hipFuncAttributeMaxDynamicSharedMemorySize, (int)shB);
  hipFuncSetAttribute((const void*)k_node_out,
                      hipFuncAttributeMaxDynamicSharedMemorySize, (int)shC);

  k_node_linear1<<<nodeTiles, THREADS, shA, stream>>>(x, w1, b1, hsum, hbuf, nNodes);
  // ~1 WG per WGP (LDS-bound); each WG loops over ~(12500/768)=16 edge tiles.
  k_edge_mlp<<<768, THREADS, shB, stream>>>(ei, ea, xp, mw1, mb1, mw2, mb2,
                                            mw3, mb3, hbuf, hsum, E);
  k_node_out<<<nodeTiles, THREADS, shC, stream>>>(hsum, w2, b2, w3, b3,
                                                  (float*)d_out, nNodes);
}
